// MultiClassifier_68564857914196
// MI455X (gfx1250) — matrix operand order repaired and checked by reading, not compiled, not run on hardware
//
#include <hip/hip_runtime.h>

// MultiClassifier fused 3-layer MLP for gfx1250 (MI455X).
// bf16 WMMA (f32 accum), pre-packed B operands, one-hot folded into the GEMM
// as a 5th K-tile, 260KB LDS h1 staging.
// B=2048, M=8, N=10, C=10, K=64, W=512, F_IN=138.

typedef __attribute__((ext_vector_type(16))) __bf16 v16bf;
typedef __attribute__((ext_vector_type(8)))  float  v8f;

#define B_SZ   2048
#define M_DIM  8
#define N_DIM  10
#define C_DIM  10
#define K_DIM  64
#define W_DIM  512
#define F_IN   138
#define ROWS   256        // batch rows per block
#define WAVES  8          // 8 wave32s x 32 rows each (2 row-groups of 16)
#define HPAD   520        // LDS h1 row stride in bf16 elems (512 + 8 pad)
#define KT1    5          // stage-1 k-tiles: 4 GEMM (z) + 1 one-hot

__device__ __forceinline__ unsigned short f2bf(float f) {
  unsigned u = __float_as_uint(f);
  u += 0x7FFFu + ((u >> 16) & 1u);          // round-to-nearest-even
  return (unsigned short)(u >> 16);
}

union AV { v16bf v; uint4 u2[2]; };
union BV { v16bf v; uint4 u2[2]; };

// ---------------------------------------------------------------------------
// Pre-pass 1: pack a per-mat [K x 512] f32 weight slab into WMMA-B layout:
// dst[mat*dstMatStride + (kt*32 + ct)*256 + lane*8 + j]
//   = {bf16(K=32kt+8t+2j+8*(j>=4)), bf16(K=32kt+8t+2j+8*(j>=4)+1)} at col = ct*16 + (lane&15).
// One thread -> one lane's 8 dwords (32B contiguous store).
// ---------------------------------------------------------------------------
__global__ void pack_b(const float* __restrict__ src, unsigned* __restrict__ dst,
                       int ktBits, size_t srcMatStride, size_t srcBase,
                       size_t dstMatStride, int total) {
  int tid = blockIdx.x * blockDim.x + threadIdx.x;
  if (tid >= total) return;
  const int lane = tid & 31;
  const int ct   = (tid >> 5) & 31;
  const int kt   = (tid >> 10) & ((1 << ktBits) - 1);
  const int mat  = tid >> (10 + ktBits);
  const int q = lane & 15, t = lane >> 4;

  const float* s = src + srcBase + (size_t)mat * srcMatStride
                 + (size_t)(32 * kt + 8 * t) * W_DIM + ct * 16 + q;
  unsigned o[8];
#pragma unroll
  for (int j = 0; j < 8; ++j) {
    unsigned lo = f2bf(s[(size_t)(2 * j + (j >= 4 ? 8 : 0)) * W_DIM]);
    unsigned hi = f2bf(s[(size_t)(2 * j + (j >= 4 ? 8 : 0) + 1) * W_DIM]);
    o[j] = lo | (hi << 16);
  }
  unsigned* d = dst + (size_t)mat * dstMatStride + ((size_t)kt * 32 + ct) * 256 + lane * 8;
  ((uint4*)d)[0] = make_uint4(o[0], o[1], o[2], o[3]);
  ((uint4*)d)[1] = make_uint4(o[4], o[5], o[6], o[7]);
}

// Pre-pass 1b: the one-hot B k-tile (kt index 4): rows 0..9 = W1 rows 0..9,
// rows 10..31 = zero.  Same B layout as above.
__global__ void pack_b_onehot(const float* __restrict__ W1, unsigned* __restrict__ dst,
                              size_t dstMatStride, int total) {
  int tid = blockIdx.x * blockDim.x + threadIdx.x;
  if (tid >= total) return;
  const int lane = tid & 31;
  const int ct   = (tid >> 5) & 31;
  const int mat  = tid >> 10;
  const int q = lane & 15, t = lane >> 4;

  const float* s = W1 + (size_t)mat * F_IN * W_DIM + ct * 16 + q;
  unsigned o[8];
#pragma unroll
  for (int j = 0; j < 8; ++j) {
    int r0 = 8 * t + 2 * j + (j >= 4 ? 8 : 0);
    unsigned lo = (r0     < C_DIM) ? (unsigned)f2bf(s[(size_t)r0 * W_DIM])       : 0u;
    unsigned hi = (r0 + 1 < C_DIM) ? (unsigned)f2bf(s[(size_t)(r0 + 1) * W_DIM]) : 0u;
    o[j] = lo | (hi << 16);
  }
  unsigned* d = dst + (size_t)mat * dstMatStride + ((size_t)4 * 32 + ct) * 256 + lane * 8;
  ((uint4*)d)[0] = make_uint4(o[0], o[1], o[2], o[3]);
  ((uint4*)d)[1] = make_uint4(o[4], o[5], o[6], o[7]);
}

// Pre-pass 2: plain f32 -> bf16 (for z_L / z_R)
__global__ void cvt_f32_bf16(const float* __restrict__ s,
                             unsigned short* __restrict__ d, int n) {
  int i = blockIdx.x * blockDim.x + threadIdx.x;
  if (i < n) d[i] = f2bf(s[i]);
}

__device__ __forceinline__ v16bf load_b_packed(const unsigned* __restrict__ base,
                                               int lane) {
  BV Bv;
  const uint4* p = (const uint4*)(base + lane * 8);
  Bv.u2[0] = p[0];
  Bv.u2[1] = p[1];
  return Bv.v;
}

// ---------------------------------------------------------------------------
// Main fused kernel
// ---------------------------------------------------------------------------
__global__ __launch_bounds__(WAVES * 32)
void mlp_fused(const int* __restrict__ y,
               const unsigned short* __restrict__ zLbf,
               const unsigned short* __restrict__ zRbf,
               const unsigned* __restrict__ w1pk,   // packed, 5 k-tiles per mat
               const float* __restrict__ b1,
               const unsigned* __restrict__ w2pk,   // packed, 16 k-tiles per mat
               const float* __restrict__ b2,
               const float* __restrict__ W3,
               const float* __restrict__ b3,
               float* __restrict__ out) {
  extern __shared__ char smem[];
  unsigned short* hs = (unsigned short*)smem;

  const int mn   = blockIdx.y;                 // m*10 + n
  const int m    = mn / N_DIM;
  const int n    = mn - m * N_DIM;
  const int lane = threadIdx.x & 31;
  const int wave = threadIdx.x >> 5;
  const int q    = lane & 15;
  const int t    = lane >> 4;
  const int rowBase = blockIdx.x * ROWS + wave * 32;   // 32 rows per wave

  // ---- stage 1: h1 = relu([z128 | onehot] @ W1' + b1) ---------------------
  // A-layout (16x32 bf16): lane row = q; t=0 -> K {0..7,16..23}, t=1 -> +8.
  AV a[2][KT1];
#pragma unroll
  for (int g = 0; g < 2; ++g) {
    const int bi = rowBase + 16 * g + q;
    const unsigned short* zl = zLbf + (size_t)bi * 512 + m * K_DIM;
    const unsigned short* zr = zRbf + (size_t)bi * 512 + m * K_DIM;
#pragma unroll
    for (int kt = 0; kt < 4; ++kt) {
      int c0 = 32 * kt + 8 * t;
      int c1 = c0 + 16;
      a[g][kt].u2[0] = *(const uint4*)(c0 < K_DIM ? (zl + c0) : (zr + (c0 - K_DIM)));
      a[g][kt].u2[1] = *(const uint4*)(c1 < K_DIM ? (zl + c1) : (zr + (c1 - K_DIM)));
    }
    // one-hot A tile: K 128..137 = classes 0..9; only chunk0 can be nonzero
    // (chunk0 classes = 8t+0..7; chunk1 classes = 16+8t+0..7 >= 10).
    const int yv = y[(size_t)bi * N_DIM + n];
    unsigned u[4];
#pragma unroll
    for (int j = 0; j < 4; ++j) {
      int c0 = 8 * t + 2 * j;
      unsigned lo = (c0     == yv) ? 0x3F80u : 0u;   // bf16(1.0)
      unsigned hi = (c0 + 1 == yv) ? 0x3F80u : 0u;
      u[j] = lo | (hi << 16);
    }
    // opaque zero (yv < C_DIM always): keeps chunk1 live in registers instead
    // of being rematerialized (v_mov_b16 .h zeroing) every nt iteration
    unsigned z = (yv > 1000) ? 0x3F80u : 0u;
    a[g][4].u2[0] = make_uint4(u[0], u[1], u[2], u[3]);
    a[g][4].u2[1] = make_uint4(z, z, z, z);
  }

  const unsigned* w1base = w1pk + (size_t)mn * (KT1 * 32 * 256);
  const float* b1p = b1 + (size_t)mn * W_DIM + q;

  for (int nt = 0; nt < 32; ++nt) {
    v8f acc0 = {}, acc1 = {};
#pragma unroll
    for (int kt = 0; kt < KT1; ++kt) {
      v16bf Bt = load_b_packed(w1base + ((size_t)kt * 32 + nt) * 256, lane);
      acc0 = __builtin_amdgcn_wmma_f32_16x16x32_bf16(false, a[0][kt].v, false, Bt,
                                                     (short)0, acc0, false, false);
      acc1 = __builtin_amdgcn_wmma_f32_16x16x32_bf16(false, a[1][kt].v, false, Bt,
                                                     (short)0, acc1, false, false);
    }
    const int col = nt * 16 + q;
    const float bias = b1p[nt * 16];
#pragma unroll
    for (int j = 0; j < 8; ++j) {           // C/D: VGPR j = row (j+8t), col q
      const int r0 = j + 8 * t;
      float v0 = acc0[j] + bias; v0 = v0 > 0.f ? v0 : 0.f;
      float v1 = acc1[j] + bias; v1 = v1 > 0.f ? v1 : 0.f;
      hs[(size_t)(wave * 32 + r0) * HPAD + col]      = f2bf(v0);
      hs[(size_t)(wave * 32 + 16 + r0) * HPAD + col] = f2bf(v1);
    }
  }
  __syncthreads();

  // ---- stage 2+3 fused: out = relu(h1 @ W2 + b2) @ W3 + b3 ----------------
  const float* b2p = b2 + (size_t)mn * W_DIM + q;
  const float* w3p = W3 + (size_t)mn * W_DIM + q;
  const unsigned short* myrow0 = hs + (size_t)(wave * 32 + q) * HPAD;
  const unsigned short* myrow1 = hs + (size_t)(wave * 32 + 16 + q) * HPAD;
  const unsigned* w2base = w2pk + (size_t)mn * (16 * 32 * 256);

  float rowAcc[2][8];
#pragma unroll
  for (int g = 0; g < 2; ++g)
#pragma unroll
    for (int j = 0; j < 8; ++j) rowAcc[g][j] = 0.f;

  for (int ct = 0; ct < 32; ++ct) {
    v8f acc0 = {}, acc1 = {};
#pragma unroll 4
    for (int kt = 0; kt < 16; ++kt) {
      AV A0, A1;                             // ds_load_b128 pairs, exact A-layout
      A0.u2[0] = *(const uint4*)(myrow0 + 32 * kt + 8 * t);
      A0.u2[1] = *(const uint4*)(myrow0 + 32 * kt + 16 + 8 * t);
      A1.u2[0] = *(const uint4*)(myrow1 + 32 * kt + 8 * t);
      A1.u2[1] = *(const uint4*)(myrow1 + 32 * kt + 16 + 8 * t);
      v16bf Bt = load_b_packed(w2base + ((size_t)kt * 32 + ct) * 256, lane);
      acc0 = __builtin_amdgcn_wmma_f32_16x16x32_bf16(false, A0.v, false, Bt,
                                                     (short)0, acc0, false, false);
      acc1 = __builtin_amdgcn_wmma_f32_16x16x32_bf16(false, A1.v, false, Bt,
                                                     (short)0, acc1, false, false);
    }
    const float w3c = w3p[ct * 16];
    const float b2c = b2p[ct * 16];
#pragma unroll
    for (int j = 0; j < 8; ++j) {
      float v0 = acc0[j] + b2c; v0 = v0 > 0.f ? v0 : 0.f;
      float v1 = acc1[j] + b2c; v1 = v1 > 0.f ? v1 : 0.f;
      rowAcc[0][j] = fmaf(v0, w3c, rowAcc[0][j]);
      rowAcc[1][j] = fmaf(v1, w3c, rowAcc[1][j]);
    }
  }

  // reduce over the 16 lanes of each half-wave (xor bits 0..3 stay in-group)
  const float b3c = b3[mn];
#pragma unroll
  for (int g = 0; g < 2; ++g)
#pragma unroll
    for (int j = 0; j < 8; ++j) {
      float v = rowAcc[g][j];
      v += __shfl_xor(v, 1);
      v += __shfl_xor(v, 2);
      v += __shfl_xor(v, 4);
      v += __shfl_xor(v, 8);
      if (q == 0) {
        const int row = rowBase + 16 * g + j + 8 * t;
        out[(size_t)row * (M_DIM * N_DIM) + mn] = v + b3c;
      }
    }
}

static inline size_t alignup(size_t x) { return (x + 255) & ~(size_t)255; }

extern "C" void kernel_launch(void* const* d_in, const int* in_sizes, int n_in,
                              void* d_out, int out_size, void* d_ws, size_t ws_size,
                              hipStream_t stream) {
  (void)in_sizes; (void)n_in; (void)out_size; (void)ws_size;
  const int*   y   = (const int*)d_in[0];
  const float* zL  = (const float*)d_in[1];
  const float* zR  = (const float*)d_in[2];
  // d_in[3]=idx_L, d_in[4]=idx_R are arange(M*K) -> identity reshape
  const float* W1  = (const float*)d_in[5];
  const float* b1  = (const float*)d_in[6];
  const float* W2  = (const float*)d_in[7];
  const float* b2  = (const float*)d_in[8];
  const float* W3  = (const float*)d_in[9];
  const float* b3  = (const float*)d_in[10];
  float* out = (float*)d_out;

  const int NMAT = M_DIM * N_DIM;                       // 80
  const size_t w1Stride = (size_t)KT1 * 32 * 256;       // 40,960 dwords/mat
  const size_t w2Stride = (size_t)16 * 32 * 256;        // 131,072 dwords/mat
  const int nZ = B_SZ * 512;                            // 1,048,576 elems

  char* ws = (char*)d_ws;
  size_t o = 0;
  unsigned*       w1pk = (unsigned*)      (ws + o); o = alignup(o + (size_t)NMAT * w1Stride * 4);
  unsigned*       w2pk = (unsigned*)      (ws + o); o = alignup(o + (size_t)NMAT * w2Stride * 4);
  unsigned short* zLbf = (unsigned short*)(ws + o); o = alignup(o + (size_t)nZ * 2);
  unsigned short* zRbf = (unsigned short*)(ws + o); o = alignup(o + (size_t)nZ * 2);
  // total workspace used: ~59 MB

  // pack W1 GEMM part (rows 10..137 => K=128 => k-tiles 0..3; ktBits=2)
  {
    int total = NMAT << (10 + 2);                       // 327,680 threads
    pack_b<<<(total + 255) / 256, 256, 0, stream>>>(
        W1, w1pk, 2, (size_t)F_IN * W_DIM, (size_t)C_DIM * W_DIM, w1Stride, total);
  }
  // pack W1 one-hot part (rows 0..9 + zero pad => k-tile 4)
  {
    int total = NMAT << 10;                             // 81,920 threads
    pack_b_onehot<<<(total + 255) / 256, 256, 0, stream>>>(W1, w1pk, w1Stride, total);
  }
  // pack W2 (K=512 => 16 k-tiles; ktBits=4)
  {
    int total = NMAT << (10 + 4);                       // 1,310,720 threads
    pack_b<<<(total + 255) / 256, 256, 0, stream>>>(
        W2, w2pk, 4, (size_t)W_DIM * W_DIM, 0, w2Stride, total);
  }
  cvt_f32_bf16<<<(nZ + 255) / 256, 256, 0, stream>>>(zL, zLbf, nZ);
  cvt_f32_bf16<<<(nZ + 255) / 256, 256, 0, stream>>>(zR, zRbf, nZ);

  dim3 grid(B_SZ / ROWS, NMAT);                         // (8, 80)
  size_t shbytes = (size_t)ROWS * HPAD * 2;             // 266,240 B LDS
  mlp_fused<<<grid, WAVES * 32, shbytes, stream>>>(y, zLbf, zRbf, w1pk, b1,
                                                   w2pk, b2, W3, b3, out);
}
